// IsingRBM_35021163331997
// MI455X (gfx1250) — compile-verified, matrix operand order checked by reading, not run on hardware
//
#include <hip/hip_runtime.h>

typedef __attribute__((ext_vector_type(16))) _Float16 v16h;
typedef __attribute__((ext_vector_type(8)))  _Float16 v8h;
typedef __attribute__((ext_vector_type(8)))  float    v8f;

// Problem constants (fixed by the reference)
#define NQ   64
#define HH   64
#define NBAT 32768

// ---------------------------------------------------------------------------
// Pre-pass: convert W1 (64x64, [a][h]) and W2 (64x64x64, [h][a][b]) to f16 and
// swizzle them into per-lane WMMA B-fragment layout.
//
// B-fragment (16-bit, 32x16, v16h per lane), mirroring the documented A layout:
//   lane<16 : column n = lane,    K halves = {0..7, 16..23}
//   lane>=16: column n = lane-16, K halves = {8..15, 24..31}
// Element i of v16h holds K = (i&7) + 16*(i>>3) + 8*(lane>>4).
//
// bufW2: 4 h-tiles x 128 k-chunks x 32 lanes x 16 halves  (512 KB)
// bufW1: 4 h-tiles x   2 k-chunks x 32 lanes x 16 halves  (  8 KB)
// ---------------------------------------------------------------------------
__global__ void __launch_bounds__(256)
rbm_swizzle(const float* __restrict__ w1,
            const float* __restrict__ w2,
            _Float16* __restrict__ bufW1,
            _Float16* __restrict__ bufW2) {
  int tid = blockIdx.x * blockDim.x + threadIdx.x;
  if (tid >= (512 + 8) * 32) return;
  int lane  = tid & 31;
  int t     = tid >> 5;
  int khalf = lane >> 4;
  int col   = lane & 15;

  v16h v;
  if (t < 512) {                       // W2 fragments
    int c = t >> 7, q = t & 127;       // h-tile, k-chunk
    int n = c * 16 + col;              // hidden unit (column of B)
    const float* src = w2 + (size_t)n * 4096 + q * 32;  // W2[h][k], k=a*64+b
#pragma unroll
    for (int i = 0; i < 16; ++i) {
      int kk = (i & 7) + ((i >> 3) << 4) + (khalf << 3);
      v[i] = (_Float16)src[kk];
    }
    *(v16h*)(bufW2 + (size_t)tid * 16) = v;
  } else {                             // W1 fragments
    int t2 = t - 512;
    int c = t2 >> 1, q = t2 & 1;
    int n = c * 16 + col;
#pragma unroll
    for (int i = 0; i < 16; ++i) {
      int kk = (i & 7) + ((i >> 3) << 4) + (khalf << 3);
      v[i] = (_Float16)w1[(q * 32 + kk) * 64 + n];      // W1[a][h]
    }
    *(v16h*)(bufW1 + (size_t)(tid - 512 * 32) * 16) = v;
  }
}

// ---------------------------------------------------------------------------
// Main kernel: 256 threads = 8 waves. Block owns 32 samples; wave (s,c) owns
// sample sub-tile s (16 rows) x hidden tile c (16 cols).
//   acc  = x_tile @ W1_tile                      (2  x wmma_f32_16x16x32_f16)
//   acc += Z_tile @ W2T_tile, Z built on the fly (128x wmma_f32_16x16x32_f16)
//   out  = prod_h cos(acc + bias)                (butterfly product)
// ---------------------------------------------------------------------------
__global__ void __launch_bounds__(256)
rbm_main(const float* __restrict__ x,
         const float* __restrict__ hbias,
         const _Float16* __restrict__ bufW1,
         const _Float16* __restrict__ bufW2,
         float* __restrict__ out) {
  __shared__ _Float16 xh[32 * 72];   // 32 rows x 64 cols f16, padded to 72
  __shared__ float    part[32 * 4];  // per-(row, h-tile) partial products

  const int tid = threadIdx.x;

  // Stage the 32x64 f32 x-tile into LDS as f16 (coalesced float4 loads).
  const float* xblk = x + (size_t)blockIdx.x * (32 * 64);
  for (int i = tid; i < (32 * 64) / 4; i += 256) {
    float4 v = ((const float4*)xblk)[i];
    int row = i >> 4;
    int c4  = (i & 15) << 2;
    _Float16* d = &xh[row * 72 + c4];
    d[0] = (_Float16)v.x; d[1] = (_Float16)v.y;
    d[2] = (_Float16)v.z; d[3] = (_Float16)v.w;
  }
  __syncthreads();

  const int lane  = tid & 31;
  const int wv    = tid >> 5;
  const int s     = wv & 1;     // sample sub-tile (m base = 16*s)
  const int c     = wv >> 1;    // hidden tile (h base = 16*c)
  const int mloc  = lane & 15;
  const int khalf = lane >> 4;
  const _Float16* xrow = &xh[(s * 16 + mloc) * 72];

  // A fragments of x for K-chunks b=0..31 (ax0) and b=32..63 (ax1).
  // 16-bit A layout: lane<16 K={0..7,16..23}, lane>=16 K={8..15,24..31};
  // both are two 8-half contiguous runs -> ds_load_b128 pairs (16B aligned).
  v16h ax0, ax1;
  {
    v8h l0 = *(const v8h*)(xrow + 8 * khalf);
    v8h h0 = *(const v8h*)(xrow + 16 + 8 * khalf);
    v8h l1 = *(const v8h*)(xrow + 32 + 8 * khalf);
    v8h h1 = *(const v8h*)(xrow + 48 + 8 * khalf);
#pragma unroll
    for (int i = 0; i < 8; ++i) {
      ax0[i] = l0[i]; ax0[i + 8] = h0[i];
      ax1[i] = l1[i]; ax1[i + 8] = h1[i];
    }
  }

  v8f acc = {};

  // Linear term: x @ W1 over K=64 (2 chunks).
  const v16h* bw1 = (const v16h*)bufW1;
  acc = __builtin_amdgcn_wmma_f32_16x16x32_f16(
      false, ax0, false, bw1[(c * 2 + 0) * 32 + lane], (short)0, acc, false, false);
  acc = __builtin_amdgcn_wmma_f32_16x16x32_f16(
      false, ax1, false, bw1[(c * 2 + 1) * 32 + lane], (short)0, acc, false, false);

  // Quadratic term: Z @ W2T over K=4096 (128 chunks). Within chunk q the
  // outer index a = q>>1 is constant, so A(chunk) = ax[q&1] * (0.5*x[m,a]).
  const v16h* bw2 = (const v16h*)bufW2 + (size_t)c * 128 * 32 + lane;
#pragma unroll 4
  for (int q = 0; q < 128; ++q) {
    __builtin_prefetch((const void*)(bw2 + (q + 8) * 32), 0, 1);  // global_prefetch_b8
    _Float16 xah = (_Float16)((float)xrow[q >> 1] * 0.5f);        // folds the 0.5
    v16h az = ((q & 1) ? ax1 : ax0) * xah;                        // 8x v_pk_mul_f16
    acc = __builtin_amdgcn_wmma_f32_16x16x32_f16(
        false, az, false, bw2[(size_t)q * 32], (short)0, acc, false, false);
  }

  // Epilogue: cos, then product over the 16 hidden units of this tile.
  // C layout: VGPR j / lane -> pre[m = j + 8*khalf, h = 16*c + mloc].
  float bias = hbias[c * 16 + mloc];
#pragma unroll
  for (int j = 0; j < 8; ++j) {
    float v = __builtin_cosf(acc[j] + bias);
    v *= __shfl_xor(v, 1);   // butterfly product within each 16-lane group
    v *= __shfl_xor(v, 2);
    v *= __shfl_xor(v, 4);
    v *= __shfl_xor(v, 8);
    if (mloc == 0) part[(s * 16 + j + 8 * khalf) * 4 + c] = v;
  }
  __syncthreads();

  // Combine the 4 hidden tiles per sample row and write out.
  if (tid < 32) {
    float pr = part[tid * 4 + 0] * part[tid * 4 + 1] *
               part[tid * 4 + 2] * part[tid * 4 + 3];
    out[(size_t)blockIdx.x * 32 + tid] = pr;
  }
}

extern "C" void kernel_launch(void* const* d_in, const int* in_sizes, int n_in,
                              void* d_out, int out_size, void* d_ws, size_t ws_size,
                              hipStream_t stream) {
  const float* x     = (const float*)d_in[0];  // (32768, 64)
  const float* w1    = (const float*)d_in[1];  // (64, 64)
  const float* w2    = (const float*)d_in[2];  // (64, 64, 64)
  const float* hbias = (const float*)d_in[3];  // (64,)
  float* out = (float*)d_out;                  // (32768,)

  _Float16* bufW2 = (_Float16*)d_ws;           // 512 KB
  _Float16* bufW1 = bufW2 + 512 * 32 * 16;     //   8 KB

  rbm_swizzle<<<65, 256, 0, stream>>>(w1, w2, bufW1, bufW2);
  rbm_main<<<NBAT / 32, 256, 0, stream>>>(x, hbias, bufW1, bufW2, out);
}
